// MultiHeadAttention_62921270886433
// MI455X (gfx1250) — compile-verified, matrix operand order checked by reading, not run on hardware
//
#include <hip/hip_runtime.h>
#include <hip/hip_bf16.h>
#include <cstddef>
#include <cstdint>

// ---------------------------------------------------------------------------
// MHA forward for MI455X (gfx1250), wave32, WMMA bf16 path.
//   D_MODEL=1024, H=16, d_k=64, N=2, L=2048
// Pipeline:
//   1) convert q,k,v,W_q,W_k,W_v,W_o fp32 -> bf16 (workspace)
//   2) Q/K/V projections: bf16 WMMA GEMM  C = X @ W^T + b   (bf16 out)
//      tiles staged with GLOBAL_LOAD_ASYNC_TO_LDS_B128 (+ s_wait_asynccnt)
//   3) flash attention per (n,h): sweep1 online softmax + O=P@V (WMMA),
//      sweep2 recompute S (WMMA) and write normalized attn fp32 ONCE
//   4) out = ctx @ W_o^T + b_o (fp32 out)
// ---------------------------------------------------------------------------

typedef __attribute__((ext_vector_type(16))) __bf16 v16bf;
typedef __attribute__((ext_vector_type(8)))  __bf16 v8bf;
typedef __attribute__((ext_vector_type(4)))  __bf16 v4bf;
typedef __attribute__((ext_vector_type(8)))  float  v8f;

#define D_MODEL 1024
#define NHEADS  16
#define DK      64
#define LSEQ    2048
#define NB      2

__device__ __forceinline__ v8f wmma_bf16(v16bf a, v16bf b, v8f c) {
  return __builtin_amdgcn_wmma_f32_16x16x32_bf16(
      /*neg_a=*/false, a, /*neg_b=*/false, b,
      /*c_mod=*/(short)0, c, /*reuse_a=*/false, /*reuse_b=*/false);
}

// Async global->LDS copy of 16 bytes (CDNA5 ASYNCcnt path). The LDS address
// is the low 32 bits of the generic pointer;
// the global address is a 64-bit VGPR pair.
__device__ __forceinline__ void async_ld_b128(void* lds, const void* gptr) {
  asm volatile("global_load_async_to_lds_b128 %0, %1, off"
               :
               : "v"((unsigned)(uintptr_t)lds), "v"((uint64_t)(uintptr_t)gptr)
               : "memory");
}
__device__ __forceinline__ void wait_async0() {
  asm volatile("s_wait_asynccnt 0" ::: "memory");
}

// Load one 16x32 bf16 WMMA A/B fragment from an LDS tile stored row-major
// with `stride` bf16 per row.  Per the CDNA5 ISA layout, lane L holds
// row (L&15); lanes 0-15 carry K = {0..7, 16..23}, lanes 16-31 carry
// K = {8..15, 24..31}.  r = lane&15, kb = (lane>>4)*8.
__device__ __forceinline__ v16bf load_frag(const __bf16* base, int stride,
                                           int r, int kb) {
  const __bf16* p = base + r * stride;
  v8bf lo = *(const v8bf*)(p + kb);
  v8bf hi = *(const v8bf*)(p + 16 + kb);
  v16bf f;
#pragma unroll
  for (int i = 0; i < 8; ++i) { f[i] = lo[i]; f[i + 8] = hi[i]; }
  return f;
}

// ---------------------------------------------------------------------------
// fp32 -> bf16 conversion (vectorized x4)
// ---------------------------------------------------------------------------
__global__ void cvt_f32_bf16(const float* __restrict__ in,
                             __bf16* __restrict__ out, int n) {
  int i = (blockIdx.x * blockDim.x + threadIdx.x) * 4;
  if (i + 3 < n) {
    float4 v = *(const float4*)(in + i);
    v4bf o;
    o[0] = (__bf16)v.x; o[1] = (__bf16)v.y;
    o[2] = (__bf16)v.z; o[3] = (__bf16)v.w;
    *(v4bf*)(out + i) = o;
  }
}

// ---------------------------------------------------------------------------
// GEMM: C[M,Nc] = A[M,Kd] @ W[Nc,Kd]^T + bias[Nc]
// 128x128x32 tiles, 256 threads = 8 waves, wave tile 32x64 (2x4 WMMA tiles).
// Tile staging via async global->LDS. Writes bf16 (Cb) or fp32 (Cf).
// ---------------------------------------------------------------------------
#define GBM 128
#define GBN 128
#define GBK 32
#define GKP 40   // padded K stride (bf16 elems), 80B rows keep 16B alignment

__global__ __launch_bounds__(256)
void gemm_xwT_bf16(const __bf16* __restrict__ A, const __bf16* __restrict__ W,
                   const float* __restrict__ bias,
                   __bf16* __restrict__ Cb, float* __restrict__ Cf,
                   int M, int Nc, int Kd) {
  __shared__ __bf16 As[GBM][GKP];
  __shared__ __bf16 Bs[GBN][GKP];

  const int t    = threadIdx.x;
  const int lane = t & 31;
  const int wave = t >> 5;
  const int wm   = wave >> 1;        // 0..3 : wave M position
  const int wn   = wave & 1;         // 0..1 : wave N position
  const int m0   = blockIdx.y * GBM;
  const int n0   = blockIdx.x * GBN;
  const int r    = lane & 15;
  const int kb   = (lane >> 4) * 8;
  const int hl   = lane >> 4;

  v8f acc[2][4];
#pragma unroll
  for (int i = 0; i < 2; ++i)
#pragma unroll
    for (int j = 0; j < 4; ++j) acc[i][j] = {};

  const int ldrow = t >> 1;          // 0..127
  const int ldk   = (t & 1) * 16;    // 0 or 16

  for (int k0 = 0; k0 < Kd; k0 += GBK) {
    __syncthreads();
    {
      const __bf16* sa = A + (size_t)(m0 + ldrow) * Kd + k0 + ldk;
      const __bf16* sw = W + (size_t)(n0 + ldrow) * Kd + k0 + ldk;
      async_ld_b128(&As[ldrow][ldk],     sa);
      async_ld_b128(&As[ldrow][ldk + 8], sa + 8);
      async_ld_b128(&Bs[ldrow][ldk],     sw);
      async_ld_b128(&Bs[ldrow][ldk + 8], sw + 8);
      if (k0 + GBK < Kd) {           // pull next tiles toward the WGP
        __builtin_prefetch(sa + GBK, 0, 0);
        __builtin_prefetch(sw + GBK, 0, 0);
      }
    }
    wait_async0();
    __syncthreads();

    v16bf af[2], bf_[4];
#pragma unroll
    for (int i = 0; i < 2; ++i)
      af[i] = load_frag(&As[wm * 32 + i * 16][0], GKP, r, kb);
#pragma unroll
    for (int j = 0; j < 4; ++j)
      bf_[j] = load_frag(&Bs[wn * 64 + j * 16][0], GKP, r, kb);

#pragma unroll
    for (int i = 0; i < 2; ++i)
#pragma unroll
      for (int j = 0; j < 4; ++j)
        acc[i][j] = wmma_bf16(af[i], bf_[j], acc[i][j]);
  }

  // Epilogue: C layout -> VGPR rr holds M = rr + 8*(lane>=16), N = lane&15.
#pragma unroll
  for (int i = 0; i < 2; ++i)
#pragma unroll
    for (int j = 0; j < 4; ++j)
#pragma unroll
      for (int rr = 0; rr < 8; ++rr) {
        int row = m0 + wm * 32 + i * 16 + rr + hl * 8;
        int col = n0 + wn * 64 + j * 16 + (lane & 15);
        float v = acc[i][j][rr] + bias[col];
        if (Cb) Cb[(size_t)row * Nc + col] = (__bf16)v;
        else    Cf[(size_t)row * Nc + col] = v;
      }
}

// ---------------------------------------------------------------------------
// Flash attention per (n,h). 128 queries/block (8 waves x 16 rows),
// 64-key tiles. Sweep 1: online softmax + O = P@V. Sweep 2: recompute S,
// write normalized attn fp32 (single write of the 536MB attn tensor).
// Q block and K tiles staged via async global->LDS.
// ---------------------------------------------------------------------------
#define AQ   128
#define AKT  64
#define APAD 72   // 144B rows keep 16B alignment

__global__ __launch_bounds__(256)
void mha_flash(const __bf16* __restrict__ Qp, const __bf16* __restrict__ Kp,
               const __bf16* __restrict__ Vp, const int* __restrict__ mask,
               __bf16* __restrict__ ctx, float* __restrict__ attn) {
  __shared__ __bf16 Qs[AQ][APAD];
  __shared__ __bf16 Ks[AKT][APAD];
  __shared__ __bf16 Vt[DK][APAD];        // V transposed: [d][key]
  __shared__ __bf16 Ps[8][16][APAD];     // per-wave P slab (C->A transit)
  __shared__ int    mskS[AKT];

  const int t    = threadIdx.x;
  const int lane = t & 31;
  const int wave = t >> 5;
  const int r    = lane & 15;
  const int kb   = (lane >> 4) * 8;
  const int hl   = lane >> 4;

  const int nh = blockIdx.y;
  const int n  = nh >> 4;
  const int h  = nh & 15;
  const int q0 = blockIdx.x * AQ;

  const __bf16* Qb = Qp + ((size_t)n * LSEQ) * D_MODEL + (size_t)h * DK;
  const __bf16* Kb = Kp + ((size_t)n * LSEQ) * D_MODEL + (size_t)h * DK;
  const __bf16* Vb = Vp + ((size_t)n * LSEQ) * D_MODEL + (size_t)h * DK;

  // Load Q block once: 128x64, 32 bf16 per thread (4 async b128 copies).
  {
    int row = t >> 1;
    int c   = (t & 1) * 32;
    const __bf16* src = Qb + (size_t)(q0 + row) * D_MODEL + c;
#pragma unroll
    for (int u = 0; u < 4; ++u)
      async_ld_b128(&Qs[row][c + u * 8], src + u * 8);
  }

  const float NEG_INF = -__builtin_inff();
  const float scale   = 0.125f;  // 1/sqrt(64)

  float mrow[8], lsum[8];
  v8f   Oacc[4];
#pragma unroll
  for (int i = 0; i < 8; ++i) { mrow[i] = NEG_INF; lsum[i] = 0.f; }
#pragma unroll
  for (int i = 0; i < 4; ++i) Oacc[i] = {};

  // ------------------------- sweep 1 -------------------------
  for (int kt = 0; kt < LSEQ; kt += AKT) {
    __syncthreads();
    {
      int row = t >> 2;              // 0..63
      int c   = (t & 3) * 16;        // 0,16,32,48
      const __bf16* sk = Kb + (size_t)(kt + row) * D_MODEL + c;
      async_ld_b128(&Ks[row][c],     sk);
      async_ld_b128(&Ks[row][c + 8], sk + 8);
      const __bf16* sv = Vb + (size_t)(kt + row) * D_MODEL + c;
      v8bf v0 = *(const v8bf*)(sv);
      v8bf v1 = *(const v8bf*)(sv + 8);
#pragma unroll
      for (int u = 0; u < 8; ++u) {
        Vt[c + u][row]     = v0[u];
        Vt[c + 8 + u][row] = v1[u];
      }
      if (t < AKT) mskS[t] = mask[(size_t)n * LSEQ + kt + t];
    }
    wait_async0();
    __syncthreads();

    // S = Q @ K^T : 4 key tiles of 16, d_k=64 -> 2 WMMA K-steps.
    v8f s[4];
#pragma unroll
    for (int j = 0; j < 4; ++j) s[j] = {};
#pragma unroll
    for (int ks = 0; ks < 2; ++ks) {
      v16bf a = load_frag(&Qs[wave * 16][ks * 32], APAD, r, kb);
#pragma unroll
      for (int j = 0; j < 4; ++j) {
        v16bf b = load_frag(&Ks[j * 16][ks * 32], APAD, r, kb);
        s[j] = wmma_bf16(a, b, s[j]);
      }
    }

    // scale + mask + row max
    float pm[4][8], rmax[8];
#pragma unroll
    for (int rr = 0; rr < 8; ++rr) rmax[rr] = NEG_INF;
#pragma unroll
    for (int j = 0; j < 4; ++j) {
      bool ok = mskS[j * 16 + (lane & 15)] != 0;
#pragma unroll
      for (int rr = 0; rr < 8; ++rr) {
        float v = ok ? s[j][rr] * scale : NEG_INF;
        pm[j][rr] = v;
        rmax[rr]  = fmaxf(rmax[rr], v);
      }
    }
#pragma unroll
    for (int off = 8; off >= 1; off >>= 1)
#pragma unroll
      for (int rr = 0; rr < 8; ++rr)
        rmax[rr] = fmaxf(rmax[rr], __shfl_xor(rmax[rr], off, 32));

    float alpha[8], rsum[8];
#pragma unroll
    for (int rr = 0; rr < 8; ++rr) {
      float mnew = fmaxf(mrow[rr], rmax[rr]);
      alpha[rr]  = (mnew == NEG_INF) ? 0.f : __expf(mrow[rr] - mnew);
      mrow[rr]   = mnew;
      rsum[rr]   = 0.f;
    }
#pragma unroll
    for (int j = 0; j < 4; ++j)
#pragma unroll
      for (int rr = 0; rr < 8; ++rr) {
        float p = (pm[j][rr] == NEG_INF) ? 0.f : __expf(pm[j][rr] - mrow[rr]);
        rsum[rr] += p;
        Ps[wave][rr + hl * 8][j * 16 + (lane & 15)] = (__bf16)p;
      }
#pragma unroll
    for (int off = 8; off >= 1; off >>= 1)
#pragma unroll
      for (int rr = 0; rr < 8; ++rr)
        rsum[rr] += __shfl_xor(rsum[rr], off, 32);
#pragma unroll
    for (int rr = 0; rr < 8; ++rr)
      lsum[rr] = lsum[rr] * alpha[rr] + rsum[rr];

    // rescale O, then O += P @ V
#pragma unroll
    for (int j = 0; j < 4; ++j)
#pragma unroll
      for (int rr = 0; rr < 8; ++rr)
        Oacc[j][rr] *= alpha[rr];
#pragma unroll
    for (int ks = 0; ks < 2; ++ks) {
      v16bf a = load_frag(&Ps[wave][0][ks * 32], APAD, r, kb);
#pragma unroll
      for (int j = 0; j < 4; ++j) {
        v16bf b = load_frag(&Vt[j * 16][ks * 32], APAD, r, kb);
        Oacc[j] = wmma_bf16(a, b, Oacc[j]);
      }
    }
  }

  // normalize + store context (bf16), layout [N, L, H*dk] == [N, L, D_MODEL]
  float inv[8];
#pragma unroll
  for (int rr = 0; rr < 8; ++rr)
    inv[rr] = (lsum[rr] > 0.f) ? 1.f / lsum[rr] : 0.f;
  {
    __bf16* cdst = ctx + ((size_t)n * LSEQ) * D_MODEL + (size_t)h * DK;
#pragma unroll
    for (int rr = 0; rr < 8; ++rr) {
      int row = q0 + wave * 16 + rr + hl * 8;
#pragma unroll
      for (int j = 0; j < 4; ++j) {
        int d = j * 16 + (lane & 15);
        cdst[(size_t)row * D_MODEL + d] = (__bf16)(Oacc[j][rr] * inv[rr]);
      }
    }
  }

  // ------------------------- sweep 2: write attn -------------------------
  float* adst = attn + (size_t)nh * LSEQ * LSEQ;
  for (int kt = 0; kt < LSEQ; kt += AKT) {
    __syncthreads();
    {
      int row = t >> 2;
      int c   = (t & 3) * 16;
      const __bf16* sk = Kb + (size_t)(kt + row) * D_MODEL + c;
      async_ld_b128(&Ks[row][c],     sk);
      async_ld_b128(&Ks[row][c + 8], sk + 8);
      if (t < AKT) mskS[t] = mask[(size_t)n * LSEQ + kt + t];
    }
    wait_async0();
    __syncthreads();

    v8f s[4];
#pragma unroll
    for (int j = 0; j < 4; ++j) s[j] = {};
#pragma unroll
    for (int ks = 0; ks < 2; ++ks) {
      v16bf a = load_frag(&Qs[wave * 16][ks * 32], APAD, r, kb);
#pragma unroll
      for (int j = 0; j < 4; ++j) {
        v16bf b = load_frag(&Ks[j * 16][ks * 32], APAD, r, kb);
        s[j] = wmma_bf16(a, b, s[j]);
      }
    }
#pragma unroll
    for (int j = 0; j < 4; ++j) {
      bool ok = mskS[j * 16 + (lane & 15)] != 0;
      int key = kt + j * 16 + (lane & 15);
#pragma unroll
      for (int rr = 0; rr < 8; ++rr) {
        int row = q0 + wave * 16 + rr + hl * 8;
        float v = ok ? __expf(s[j][rr] * scale - mrow[rr]) * inv[rr] : 0.f;
        adst[(size_t)row * LSEQ + key] = v;
      }
    }
  }
}

// ---------------------------------------------------------------------------
// Host-side launch
// ---------------------------------------------------------------------------
extern "C" void kernel_launch(void* const* d_in, const int* in_sizes, int n_in,
                              void* d_out, int out_size, void* d_ws,
                              size_t ws_size, hipStream_t stream) {
  const float* q   = (const float*)d_in[0];
  const float* k   = (const float*)d_in[1];
  const float* v   = (const float*)d_in[2];
  const float* W_q = (const float*)d_in[3];
  const float* b_q = (const float*)d_in[4];
  const float* W_k = (const float*)d_in[5];
  const float* b_k = (const float*)d_in[6];
  const float* W_v = (const float*)d_in[7];
  const float* b_v = (const float*)d_in[8];
  const float* W_o = (const float*)d_in[9];
  const float* b_o = (const float*)d_in[10];
  const int*   msk = (const int*)d_in[11];

  const int M    = NB * LSEQ;                 // 4096
  const int actN = M * D_MODEL;               // 4,194,304 activation elems
  const int wN   = D_MODEL * D_MODEL;         // 1,048,576 weight elems

  // workspace layout (bf16 buffers)
  char* ws = (char*)d_ws;
  size_t off = 0;
  __bf16* qb  = (__bf16*)(ws + off); off += (size_t)actN * 2;
  __bf16* kb  = (__bf16*)(ws + off); off += (size_t)actN * 2;
  __bf16* vb  = (__bf16*)(ws + off); off += (size_t)actN * 2;
  __bf16* wqb = (__bf16*)(ws + off); off += (size_t)wN * 2;
  __bf16* wkb = (__bf16*)(ws + off); off += (size_t)wN * 2;
  __bf16* wvb = (__bf16*)(ws + off); off += (size_t)wN * 2;
  __bf16* wob = (__bf16*)(ws + off); off += (size_t)wN * 2;
  __bf16* Qp  = (__bf16*)(ws + off); off += (size_t)actN * 2;
  __bf16* Kp  = (__bf16*)(ws + off); off += (size_t)actN * 2;
  __bf16* Vp  = (__bf16*)(ws + off); off += (size_t)actN * 2;
  __bf16* ctx = (__bf16*)(ws + off); off += (size_t)actN * 2;

  float* out_f  = (float*)d_out;                       // [2,2048,1024]
  float* attn_f = out_f + (size_t)NB * LSEQ * D_MODEL; // [2,16,2048,2048]

  // 1) convert to bf16
  cvt_f32_bf16<<<actN / (4 * 256), 256, 0, stream>>>(q, qb, actN);
  cvt_f32_bf16<<<actN / (4 * 256), 256, 0, stream>>>(k, kb, actN);
  cvt_f32_bf16<<<actN / (4 * 256), 256, 0, stream>>>(v, vb, actN);
  cvt_f32_bf16<<<wN / (4 * 256), 256, 0, stream>>>(W_q, wqb, wN);
  cvt_f32_bf16<<<wN / (4 * 256), 256, 0, stream>>>(W_k, wkb, wN);
  cvt_f32_bf16<<<wN / (4 * 256), 256, 0, stream>>>(W_v, wvb, wN);
  cvt_f32_bf16<<<wN / (4 * 256), 256, 0, stream>>>(W_o, wob, wN);

  // 2) projections (bf16 WMMA GEMM, bf16 outputs)
  dim3 ggrid(D_MODEL / GBN, M / GBM);  // (8, 32)
  gemm_xwT_bf16<<<ggrid, 256, 0, stream>>>(qb, wqb, b_q, Qp, nullptr,
                                           M, D_MODEL, D_MODEL);
  gemm_xwT_bf16<<<ggrid, 256, 0, stream>>>(kb, wkb, b_k, Kp, nullptr,
                                           M, D_MODEL, D_MODEL);
  gemm_xwT_bf16<<<ggrid, 256, 0, stream>>>(vb, wvb, b_v, Vp, nullptr,
                                           M, D_MODEL, D_MODEL);

  // 3) fused flash attention + single attn write
  dim3 agrid(LSEQ / AQ, NB * NHEADS);  // (16, 32)
  mha_flash<<<agrid, 256, 0, stream>>>(Qp, Kp, Vp, msk, ctx, attn_f);

  // 4) output projection (fp32 out)
  gemm_xwT_bf16<<<ggrid, 256, 0, stream>>>(ctx, wob, b_o, nullptr, out_f,
                                           M, D_MODEL, D_MODEL);
}
